// CGCNN_pred_42442866819219
// MI455X (gfx1250) — matrix operand order repaired and checked by reading, not compiled, not run on hardware
//
#include <hip/hip_runtime.h>

#define N_NODES  50000
#define N_EDGES  800000
#define N_GRAPHS 128
#define ORIG_FEA 92
#define NBR_FEA  41
#define HDIM     64
#define KDIM     192          // 3*H
#define BN_EPS   1e-5f
#define ZPAD     200          // padded z-row stride in halves (400B -> conflict-free)

typedef __attribute__((ext_vector_type(16))) _Float16 v16h;
typedef __attribute__((ext_vector_type(8)))  _Float16 v8h;
typedef __attribute__((ext_vector_type(8)))  float    v8f;

__device__ __forceinline__ float sigmoid_f(float x) { return 1.0f / (1.0f + __expf(-x)); }
__device__ __forceinline__ float softplus_f(float x) { return (x > 20.0f) ? x : log1pf(__expf(x)); }

// ---------------- embeddings ----------------
__global__ void emb_node_kernel(const float* __restrict__ x, const float* __restrict__ W,
                                const float* __restrict__ b, float* __restrict__ h) {
    int idx = blockIdx.x * blockDim.x + threadIdx.x;
    if (idx >= N_NODES * HDIM) return;
    int node = idx >> 6, col = idx & 63;
    const float* xr = x + (size_t)node * ORIG_FEA;
    float acc = b[col];
#pragma unroll 4
    for (int k = 0; k < ORIG_FEA; ++k) acc = fmaf(xr[k], W[k * HDIM + col], acc);
    h[idx] = acc;
}

__global__ void emb_edge_kernel(const float* __restrict__ ea, const float* __restrict__ W,
                                const float* __restrict__ b, _Float16* __restrict__ e16) {
    size_t idx = (size_t)blockIdx.x * blockDim.x + threadIdx.x;
    if (idx >= (size_t)N_EDGES * HDIM) return;
    int edge = (int)(idx >> 6), col = (int)(idx & 63);
    const float* er = ea + (size_t)edge * NBR_FEA;
    float acc = b[col];
#pragma unroll
    for (int k = 0; k < NBR_FEA; ++k) acc = fmaf(er[k], W[k * HDIM + col], acc);
    e16[idx] = (_Float16)acc;
}

// ---------------- pack W into per-lane B-fragment layout ----------------
// wp[((((a*2+mat)*6 + kc)*4 + n)*32 + lane)*16 + j] =
//   W[a][kc*32 + (j&7) + (j>=8 ? 16 : 0) + (lane>=16 ? 8 : 0)][n*16 + (lane&15)]
__global__ void pack_w_kernel(const float* __restrict__ Wf, const float* __restrict__ Ws,
                              _Float16* __restrict__ wp) {
    int idx = blockIdx.x * blockDim.x + threadIdx.x;   // 73728 total
    if (idx >= 3 * 2 * 6 * 4 * 32 * 16) return;
    int j = idx & 15;
    int t = idx >> 4;
    int lane = t & 31;  t >>= 5;
    int n = t & 3;      t >>= 2;
    int kc = t % 6;     t /= 6;
    int mat = t & 1;
    int a = t >> 1;
    int k    = kc * 32 + (j & 7) + ((j & 8) ? 16 : 0) + ((lane & 16) ? 8 : 0);
    int ncol = n * 16 + (lane & 15);
    const float* src = (mat ? Ws : Wf) + ((size_t)a * KDIM + k) * HDIM + ncol;
    wp[idx] = (_Float16)(*src);
}

// ---------------- per-layer prepass: h -> f16, residual init ----------------
__global__ void conv_pre_kernel(const float* __restrict__ h, _Float16* __restrict__ h16,
                                float* __restrict__ hout) {
    int idx = blockIdx.x * blockDim.x + threadIdx.x;
    if (idx >= N_NODES * HDIM) return;
    float v = h[idx];
    h16[idx] = (_Float16)v;
    hout[idx] = v;
}

// ---------------- edge message kernel (WMMA) ----------------
// 256 threads = 8 waves; block handles 128 edges; wave handles 16 edges x 64 cols.
__global__ __launch_bounds__(256) void conv_msg_kernel(
    const _Float16* __restrict__ h16, const _Float16* __restrict__ e16,
    const int* __restrict__ edge_index,
    const _Float16* __restrict__ wp_f, const _Float16* __restrict__ wp_s,
    const float* __restrict__ bfv, const float* __restrict__ bsv,
    float* __restrict__ hout) {
    __shared__ int sdst[128];
    __shared__ int ssrc[128];
    __shared__ __align__(16) _Float16 zt[128 * ZPAD];

    const int t = threadIdx.x;
    const int ebase = blockIdx.x * 128;

    if (t < 128) {
        sdst[t] = edge_index[N_EDGES + ebase + t];   // dst = row 1 (aggregation index)
        ssrc[t] = edge_index[ebase + t];             // src = row 0
    }
    __syncthreads();

    // Stage z = [h[dst] | h[src] | e] as f16; 24 x 16B chunks per row, 3072 total.
#pragma unroll
    for (int i = 0; i < 12; ++i) {
        int c = t + i * 256;
        int row = c / 24;
        int cc  = c % 24;
        int seg = cc >> 3;
        int off = (cc & 7) * 8;                      // halves
        const _Float16* src;
        if (seg == 0)      src = h16 + (size_t)sdst[row] * HDIM + off;
        else if (seg == 1) src = h16 + (size_t)ssrc[row] * HDIM + off;
        else               src = e16 + (size_t)(ebase + row) * HDIM + off;
        uint4 v = *(const uint4*)src;
        *(uint4*)&zt[row * ZPAD + cc * 8] = v;
    }
    __syncthreads();

    const int lane = t & 31;
    const int wave = t >> 5;
    const int m    = lane & 15;
    const int koff = (lane & 16) ? 8 : 0;            // A-frag K offset for upper half-wave
    const _Float16* zrow = &zt[(wave * 16 + m) * ZPAD];

    const v8f vzero = {0.f, 0.f, 0.f, 0.f, 0.f, 0.f, 0.f, 0.f};
    v8f accF[4], accS[4];
#pragma unroll
    for (int n = 0; n < 4; ++n) { accF[n] = vzero; accS[n] = vzero; }

#pragma unroll
    for (int kc = 0; kc < 6; ++kc) {
        union { v16h v; v8h h[2]; } af;
        const v8h* ap = (const v8h*)(zrow + kc * 32 + koff);
        af.h[0] = ap[0];                             // K = kc*32 + koff + 0..7
        af.h[1] = ap[2];                             // K = kc*32 + koff + 16..23
#pragma unroll
        for (int n = 0; n < 4; ++n) {
            v16h bF = *(const v16h*)(wp_f + ((size_t)(kc * 4 + n) * 32 + lane) * 16);
            v16h bS = *(const v16h*)(wp_s + ((size_t)(kc * 4 + n) * 32 + lane) * 16);
            accF[n] = __builtin_amdgcn_wmma_f32_16x16x32_f16(
                false, af.v, false, bF, (short)0, accF[n], false, false);
            accS[n] = __builtin_amdgcn_wmma_f32_16x16x32_f16(
                false, af.v, false, bS, (short)0, accS[n], false, false);
        }
    }

    // Epilogue: msg = sigmoid(f + bf) * softplus(s + bs); scatter-add to hout[dst].
#pragma unroll
    for (int n = 0; n < 4; ++n) {
        int ncol = n * 16 + m;
        float biasF = bfv[ncol];
        float biasS = bsv[ncol];
#pragma unroll
        for (int r = 0; r < 8; ++r) {
            int M = (lane & 16) ? (8 + r) : r;       // C/D layout: upper half-wave = M+8
            float f = accF[n][r] + biasF;
            float s = accS[n][r] + biasS;
            float msg = sigmoid_f(f) * softplus_f(s);
            int dstrow = sdst[wave * 16 + M];
            atomicAdd(hout + (size_t)dstrow * HDIM + ncol, msg);
        }
    }
}

// ---------------- batch norm ----------------
__global__ void bn_stats_kernel(const float* __restrict__ h, float* __restrict__ gsum,
                                float* __restrict__ gsq) {
    __shared__ float ssum[4][64];
    __shared__ float ssq[4][64];
    int t = threadIdx.x;
    int col = t & 63, rg = t >> 6;
    float s = 0.f, q = 0.f;
    int row0 = blockIdx.x * 256 + rg;
#pragma unroll 4
    for (int i = 0; i < 64; ++i) {
        int row = row0 + i * 4;
        if (row < N_NODES) {
            float v = h[(size_t)row * HDIM + col];
            s += v; q += v * v;
        }
    }
    ssum[rg][col] = s; ssq[rg][col] = q;
    __syncthreads();
    if (t < 64) {
        float S = ssum[0][t] + ssum[1][t] + ssum[2][t] + ssum[3][t];
        float Q = ssq[0][t] + ssq[1][t] + ssq[2][t] + ssq[3][t];
        atomicAdd(gsum + t, S);
        atomicAdd(gsq + t, Q);
    }
}

__global__ void bn_apply_kernel(const float* __restrict__ hout, const float* __restrict__ gsum,
                                const float* __restrict__ gsq, const float* __restrict__ gamma,
                                const float* __restrict__ beta, float* __restrict__ h,
                                int do_relu) {
    int idx = blockIdx.x * blockDim.x + threadIdx.x;
    if (idx >= N_NODES * HDIM) return;
    int col = idx & 63;
    const float invN = 1.0f / (float)N_NODES;
    float mu  = gsum[col] * invN;
    float var = gsq[col] * invN - mu * mu;
    float v = (hout[idx] - mu) * rsqrtf(var + BN_EPS) * gamma[col] + beta[col];
    if (do_relu) v = fmaxf(v, 0.0f);
    h[idx] = v;
}

// ---------------- pooling + head ----------------
__global__ void pool_kernel(const float* __restrict__ h, const int* __restrict__ batch,
                            float* __restrict__ gpool, float* __restrict__ gcnt) {
    int idx = blockIdx.x * blockDim.x + threadIdx.x;
    if (idx >= N_NODES * HDIM) return;
    int row = idx >> 6, col = idx & 63;
    int g = batch[row];
    atomicAdd(gpool + (size_t)g * HDIM + col, h[idx]);
    if (col == 0) atomicAdd(gcnt + g, 1.0f);
}

__global__ void head_mlp_kernel(const float* __restrict__ in, const float* __restrict__ W,
                                const float* __restrict__ b, float* __restrict__ out,
                                const float* __restrict__ cnt) {
    int idx = blockIdx.x * blockDim.x + threadIdx.x;
    if (idx >= N_GRAPHS * HDIM) return;
    int g = idx >> 6, c = idx & 63;
    float inv = cnt ? (1.0f / fmaxf(cnt[g], 1.0f)) : 1.0f;
    float acc = b[c];
#pragma unroll 8
    for (int k = 0; k < HDIM; ++k) acc = fmaf(in[g * HDIM + k] * inv, W[k * HDIM + c], acc);
    out[idx] = softplus_f(acc);
}

__global__ void head_out_kernel(const float* __restrict__ y2, const float* __restrict__ Wo,
                                const float* __restrict__ bo, float* __restrict__ out) {
    int g = blockIdx.x * blockDim.x + threadIdx.x;
    if (g >= N_GRAPHS) return;
    float acc = bo[0];
#pragma unroll 8
    for (int k = 0; k < HDIM; ++k) acc = fmaf(y2[g * HDIM + k], Wo[k], acc);
    out[g] = acc;
}

// ---------------- launcher ----------------
extern "C" void kernel_launch(void* const* d_in, const int* in_sizes, int n_in,
                              void* d_out, int out_size, void* d_ws, size_t ws_size,
                              hipStream_t stream) {
    (void)in_sizes; (void)n_in; (void)out_size; (void)ws_size;
    const float* x         = (const float*)d_in[0];
    const float* edge_attr = (const float*)d_in[1];
    const int*   edge_index= (const int*)d_in[2];
    const int*   batch     = (const int*)d_in[3];
    const float* W_emb1 = (const float*)d_in[4];
    const float* b_emb1 = (const float*)d_in[5];
    const float* W_emb2 = (const float*)d_in[6];
    const float* b_emb2 = (const float*)d_in[7];
    const float* Wf  = (const float*)d_in[8];
    const float* bfv = (const float*)d_in[9];
    const float* Ws  = (const float*)d_in[10];
    const float* bsv = (const float*)d_in[11];
    const float* gamma = (const float*)d_in[12];
    const float* beta  = (const float*)d_in[13];
    const float* W1 = (const float*)d_in[14];
    const float* b1 = (const float*)d_in[15];
    const float* W2 = (const float*)d_in[16];
    const float* b2 = (const float*)d_in[17];
    const float* Wo = (const float*)d_in[18];
    const float* bo = (const float*)d_in[19];
    float* out = (float*)d_out;

    char* p = (char*)d_ws;
    auto alloc = [&](size_t bytes) -> char* {
        char* r = p;
        p += (bytes + 255) & ~(size_t)255;
        return r;
    };
    float*    h_cur = (float*)   alloc((size_t)N_NODES * HDIM * 4);
    float*    h_acc = (float*)   alloc((size_t)N_NODES * HDIM * 4);
    _Float16* h16   = (_Float16*)alloc((size_t)N_NODES * HDIM * 2);
    _Float16* e16   = (_Float16*)alloc((size_t)N_EDGES * HDIM * 2);
    _Float16* wp    = (_Float16*)alloc((size_t)3 * 2 * 12288 * 2);
    float*    gsum  = (float*)   alloc(64 * 4);
    float*    gsq   = (float*)   alloc(64 * 4);
    float*    gpool = (float*)   alloc((size_t)N_GRAPHS * HDIM * 4);
    float*    gcnt  = (float*)   alloc((size_t)N_GRAPHS * 4);
    float*    y1    = (float*)   alloc((size_t)N_GRAPHS * HDIM * 4);
    float*    y2    = (float*)   alloc((size_t)N_GRAPHS * HDIM * 4);

    emb_node_kernel<<<(N_NODES * HDIM) / 256, 256, 0, stream>>>(x, W_emb1, b_emb1, h_cur);
    emb_edge_kernel<<<(N_EDGES * HDIM) / 256, 256, 0, stream>>>(edge_attr, W_emb2, b_emb2, e16);
    pack_w_kernel<<<288, 256, 0, stream>>>(Wf, Ws, wp);

    for (int a = 0; a < 3; ++a) {
        conv_pre_kernel<<<(N_NODES * HDIM) / 256, 256, 0, stream>>>(h_cur, h16, h_acc);
        conv_msg_kernel<<<N_EDGES / 128, 256, 0, stream>>>(
            h16, e16, edge_index,
            wp + (size_t)(a * 2 + 0) * 12288, wp + (size_t)(a * 2 + 1) * 12288,
            bfv + a * HDIM, bsv + a * HDIM, h_acc);
        hipMemsetAsync(gsum, 0, 64 * 4, stream);
        hipMemsetAsync(gsq, 0, 64 * 4, stream);
        bn_stats_kernel<<<196, 256, 0, stream>>>(h_acc, gsum, gsq);
        bn_apply_kernel<<<(N_NODES * HDIM) / 256, 256, 0, stream>>>(
            h_acc, gsum, gsq, gamma + a * HDIM, beta + a * HDIM, h_cur, (a < 2) ? 1 : 0);
    }

    hipMemsetAsync(gpool, 0, (size_t)N_GRAPHS * HDIM * 4, stream);
    hipMemsetAsync(gcnt, 0, (size_t)N_GRAPHS * 4, stream);
    pool_kernel<<<(N_NODES * HDIM) / 256, 256, 0, stream>>>(h_cur, batch, gpool, gcnt);
    head_mlp_kernel<<<(N_GRAPHS * HDIM) / 256, 256, 0, stream>>>(gpool, W1, b1, y1, gcnt);
    head_mlp_kernel<<<(N_GRAPHS * HDIM) / 256, 256, 0, stream>>>(y1, W2, b2, y2, nullptr);
    head_out_kernel<<<1, 128, 0, stream>>>(y2, Wo, bo, out);
}
